// SpatioTemporalGATBatched_82008105550196
// MI455X (gfx1250) — matrix operand order repaired and checked by reading, not compiled, not run on hardware
//
#include <hip/hip_runtime.h>

typedef __attribute__((ext_vector_type(16))) __bf16 v16bf;
typedef __attribute__((ext_vector_type(8)))  float  v8f;

// ---------------------------------------------------------------- helpers

__device__ __forceinline__ __bf16 cvbf(float f) { return static_cast<__bf16>(f); }

__device__ __forceinline__ float atomic_max_f32(float* addr, float val) {
  if (val >= 0.f)
    return __int_as_float(atomicMax((int*)addr, __float_as_int(val)));
  else
    return __uint_as_float(atomicMin((unsigned int*)addr, __float_as_uint(val)));
}

__device__ __forceinline__ void edge_ends(const int* __restrict__ ei, int i, int e,
                                          int& s, int& d) {
  if (i < e) { s = ei[i]; d = ei[e + i]; }
  else       { s = i - e; d = i - e; }      // appended self-loops
}

__device__ __forceinline__ float sigm(float x) { return 1.f / (1.f + expf(-x)); }

// ---------------------------------------------------------------- WMMA GEMM
// C[M,NB] = act( A[M,K] * B + bias ), bf16 inputs / f32 accumulate.
// BT=0: B is [K,NB] row-major.  BT=1: B is [NB,K] row-major (A @ B^T).
// Requirements (all met by this workload): M % 16 == 0, K % 16 == 0, NB % 16 == 0.
// One wave computes one 16x16 tile; K unrolled in 32-chunks (compile-time padding).
template<int K, int NB, bool BT>
__global__ void gemm_bf16_wmma(const float* __restrict__ A,
                               const float* __restrict__ B,
                               const float* __restrict__ bias,
                               float* __restrict__ C,
                               int M, int act)
{
  const int wave = (int)((blockIdx.x * blockDim.x + threadIdx.x) >> 5);
  constexpr int ntn = NB >> 4;
  const int mtiles = M >> 4;
  if (wave >= mtiles * ntn) return;           // wave-uniform: EXEC stays all-ones
  const int row0 = (wave / ntn) << 4;
  const int col0 = (wave % ntn) << 4;

  const int lane = threadIdx.x & 31;
  const int hlf  = lane >> 4;                 // which half of the wave
  const int l16  = lane & 15;

  const int  arow  = row0 + l16;              // A row for this lane (always < M)
  const long abase = (long)arow * K;
  const int  bcol  = col0 + l16;              // B/C column for this lane

  v8f acc = {0.f, 0.f, 0.f, 0.f, 0.f, 0.f, 0.f, 0.f};

  constexpr int KSTEPS = (K + 31) / 32;
#pragma unroll
  for (int ks = 0; ks < KSTEPS; ++ks) {
    const int kb = ks * 32;
    v16bf av, bv;

    // ---- A tile (16x32 bf16 layout): lane = row, two contiguous 8-float runs
    {
      const float4* p0 = (const float4*)(A + abase + kb + hlf * 8);
      float4 a0 = p0[0], a1 = p0[1];
      av[0] = cvbf(a0.x); av[1] = cvbf(a0.y); av[2] = cvbf(a0.z); av[3] = cvbf(a0.w);
      av[4] = cvbf(a1.x); av[5] = cvbf(a1.y); av[6] = cvbf(a1.z); av[7] = cvbf(a1.w);
    }
    if (kb + 32 <= K) {                       // folds at compile time after unroll
      const float4* p1 = (const float4*)(A + abase + kb + 16 + hlf * 8);
      float4 a2 = p1[0], a3 = p1[1];
      av[8]  = cvbf(a2.x); av[9]  = cvbf(a2.y); av[10] = cvbf(a2.z); av[11] = cvbf(a2.w);
      av[12] = cvbf(a3.x); av[13] = cvbf(a3.y); av[14] = cvbf(a3.z); av[15] = cvbf(a3.w);
    } else {                                  // K padding: zero for both half-waves
#pragma unroll
      for (int v = 8; v < 16; ++v) av[v] = cvbf(0.f);
    }

    // ---- B tile (32x16 bf16 layout): lane = column, two K runs of 8 values, same k order as A
    if (kb + 32 <= K) {
#pragma unroll
      for (int v = 0; v < 16; ++v) {
        int kk = kb + hlf * 8 + (v & 7) + ((v >> 3) << 4);
        float g = BT ? B[(long)bcol * K + kk] : B[(long)kk * NB + bcol];
        bv[v] = cvbf(g);
      }
    } else {
      // tail step: elements 8..15 cover K padding -> mask, but load from a
      // valid address so no exec-divergent branches are generated
#pragma unroll
      for (int v = 0; v < 16; ++v) {
        const float msk = (v < 8) ? 1.f : 0.f;
        int kk = kb + hlf * 8 + (v & 7);      // always < K (K % 16 == 0)
        float g = (BT ? B[(long)bcol * K + kk] : B[(long)kk * NB + bcol]) * msk;
        bv[v] = cvbf(g);
      }
    }

    acc = __builtin_amdgcn_wmma_f32_16x16x32_bf16(false, av, false, bv,
                                                  (short)0, acc, false, false);
  }

  const float bb = bias ? bias[bcol] : 0.f;
#pragma unroll
  for (int r = 0; r < 8; ++r) {
    int row = row0 + hlf * 8 + r;             // D layout: m = half*8 + reg
    float v = acc[r] + bb;
    if (act) v = fmaxf(v, 0.f);
    C[(long)row * NB + bcol] = v;
  }
}

// ---------------------------------------------------------------- GAT pieces

// e_src[n,h] = <xp[n,h,:], a_src[h,:]>, e_dst likewise
__global__ void node_scores_kernel(const float* __restrict__ xp,
                                   const float* __restrict__ asr,
                                   const float* __restrict__ ads,
                                   float* __restrict__ esb, float* __restrict__ edb,
                                   int n)
{
  int i = blockIdx.x * blockDim.x + threadIdx.x;    // (node*4 + head)
  if (i >= n * 4) return;
  int node = i >> 2, h = i & 3;
  const float4* row = (const float4*)(xp + (long)node * 256 + h * 64);
  const float4* va  = (const float4*)(asr + h * 64);
  const float4* vd  = (const float4*)(ads + h * 64);
  float s1 = 0.f, s2 = 0.f;
#pragma unroll 4
  for (int c = 0; c < 16; ++c) {
    float4 v = row[c], a = va[c], d = vd[c];
    s1 += v.x * a.x + v.y * a.y + v.z * a.z + v.w * a.w;
    s2 += v.x * d.x + v.y * d.y + v.z * d.z + v.w * d.w;
  }
  esb[i] = s1; edb[i] = s2;
}

__global__ void init_layer_kernel(float* __restrict__ segm, float* __restrict__ segs,
                                  float* __restrict__ outacc, int n)
{
  long i = (long)blockIdx.x * blockDim.x + threadIdx.x;
  if (i < (long)n * 4) { segm[i] = -__builtin_inff(); segs[i] = 0.f; }
  if (i < (long)n * 256) outacc[i] = 0.f;
}

__global__ void edge_max_kernel(const int* __restrict__ ei,
                                const float* __restrict__ esb,
                                const float* __restrict__ edb,
                                float* __restrict__ segm, int e, int et)
{
  int i = blockIdx.x * blockDim.x + threadIdx.x;
  if (i >= et) return;
  int s, d; edge_ends(ei, i, e, s, d);
  float4 es = ((const float4*)esb)[s];
  float4 ed = ((const float4*)edb)[d];
  float v[4] = {es.x + ed.x, es.y + ed.y, es.z + ed.z, es.w + ed.w};
#pragma unroll
  for (int h = 0; h < 4; ++h) {
    float w = v[h] > 0.f ? v[h] : 0.2f * v[h];      // leaky_relu(0.2)
    atomic_max_f32(&segm[d * 4 + h], w);
  }
}

__global__ void edge_exp_kernel(const int* __restrict__ ei,
                                const float* __restrict__ esb,
                                const float* __restrict__ edb,
                                const float* __restrict__ segm,
                                float* __restrict__ segs,
                                float* __restrict__ exbuf, int e, int et)
{
  int i = blockIdx.x * blockDim.x + threadIdx.x;
  if (i >= et) return;
  int s, d; edge_ends(ei, i, e, s, d);
  float4 es = ((const float4*)esb)[s];
  float4 ed = ((const float4*)edb)[d];
  float4 mm = ((const float4*)segm)[d];
  float v[4] = {es.x + ed.x, es.y + ed.y, es.z + ed.z, es.w + ed.w};
  float m[4] = {mm.x, mm.y, mm.z, mm.w};
  float4 exv;
  float* exp_ = (float*)&exv;
#pragma unroll
  for (int h = 0; h < 4; ++h) {
    float w = v[h] > 0.f ? v[h] : 0.2f * v[h];
    float ex = expf(w - m[h]);
    exp_[h] = ex;
    atomicAdd(&segs[d * 4 + h], ex);
  }
  ((float4*)exbuf)[i] = exv;
}

// out[d, h, c] += alpha[i,h] * xp[s, h, c];  thread per (edge, channel)
__global__ void edge_agg_kernel(const int* __restrict__ ei,
                                const float* __restrict__ xp,
                                const float* __restrict__ exbuf,
                                const float* __restrict__ segs,
                                float* __restrict__ outacc, int e, int et)
{
  long t = (long)blockIdx.x * blockDim.x + threadIdx.x;
  if (t >= (long)et * 64) return;
  int i = (int)(t >> 6);
  int c = (int)(t & 63);
  int s, d; edge_ends(ei, i, e, s, d);
  float4 exv = ((const float4*)exbuf)[i];
  float4 ssv = ((const float4*)segs)[d];
  float alpha[4] = {exv.x / ssv.x, exv.y / ssv.y, exv.z / ssv.z, exv.w / ssv.w};
#pragma unroll
  for (int h = 0; h < 4; ++h) {
    atomicAdd(&outacc[(long)d * 256 + h * 64 + c],
              xp[(long)s * 256 + h * 64 + c] * alpha[h]);
  }
}

// mean over heads + bias, then LayerNorm (+ optional ReLU). One wave per node.
__global__ void headmean_ln_kernel(const float* __restrict__ outacc,
                                   const float* __restrict__ bias,
                                   const float* __restrict__ g,
                                   const float* __restrict__ beta,
                                   float* __restrict__ hout, int n, int relu)
{
  int wave = (int)((blockIdx.x * blockDim.x + threadIdx.x) >> 5);
  int lane = threadIdx.x & 31;
  if (wave >= n) return;
  const float2* row = (const float2*)(outacc + (long)wave * 256);
  float2 h0 = row[lane], h1 = row[32 + lane], h2 = row[64 + lane], h3 = row[96 + lane];
  float2 bb = ((const float2*)bias)[lane];
  float v[2];
  v[0] = 0.25f * (h0.x + h1.x + h2.x + h3.x) + bb.x;
  v[1] = 0.25f * (h0.y + h1.y + h2.y + h3.y) + bb.y;
  float sum = v[0] + v[1];
  float sq  = v[0] * v[0] + v[1] * v[1];
  for (int off = 16; off > 0; off >>= 1) {
    sum += __shfl_xor(sum, off, 32);
    sq  += __shfl_xor(sq,  off, 32);
  }
  float mu   = sum * (1.f / 64.f);
  float var  = sq * (1.f / 64.f) - mu * mu;
  float rstd = rsqrtf(var + 1e-5f);
  float2 gg = ((const float2*)g)[lane];
  float2 be = ((const float2*)beta)[lane];
  float o0 = (v[0] - mu) * rstd * gg.x + be.x;
  float o1 = (v[1] - mu) * rstd * gg.y + be.y;
  if (relu) { o0 = fmaxf(o0, 0.f); o1 = fmaxf(o1, 0.f); }
  ((float2*)(hout + (long)wave * 64))[lane] = make_float2(o0, o1);
}

// ---------------------------------------------------------------- GRU / decoder

// h0 == 0 -> gh = gru_b_hh ; temporal = (1-z)*tanh(gi_n + r*gh_n)
__global__ void gru_kernel(const float* __restrict__ gi, const float* __restrict__ bhh,
                           float* __restrict__ tmpo, int n)
{
  long i = (long)blockIdx.x * blockDim.x + threadIdx.x;
  if (i >= (long)n * 64) return;
  int node = (int)(i >> 6), c = (int)(i & 63);
  const float* g = gi + (long)node * 192;
  float r    = sigm(g[c] + bhh[c]);
  float z    = sigm(g[64 + c] + bhh[64 + c]);
  float cand = tanhf(g[128 + c] + r * bhh[128 + c]);
  tmpo[i] = (1.f - z) * cand;
}

__global__ void concat_kernel(const float* __restrict__ t, const float* __restrict__ x,
                              float* __restrict__ dcat, int n)
{
  long i = (long)blockIdx.x * blockDim.x + threadIdx.x;
  if (i >= (long)n * 80) return;
  int node = (int)(i / 80), c = (int)(i % 80);
  dcat[i] = (c < 64) ? t[(long)node * 64 + c] : x[(long)node * 16 + (c - 64)];
}

__global__ void dec2_kernel(const float* __restrict__ h, const float* __restrict__ w,
                            const float* __restrict__ b, float* __restrict__ out, int n)
{
  int node = blockIdx.x * blockDim.x + threadIdx.x;
  if (node >= n) return;
  float s0 = b[0], s1 = b[1];
  const float4* hp = (const float4*)(h + (long)node * 32);
#pragma unroll
  for (int k4 = 0; k4 < 8; ++k4) {
    float4 v = hp[k4];
    const float* wk = w + k4 * 8;
    s0 += v.x * wk[0] + v.y * wk[2] + v.z * wk[4] + v.w * wk[6];
    s1 += v.x * wk[1] + v.y * wk[3] + v.z * wk[5] + v.w * wk[7];
  }
  out[node * 2]     = fminf(fmaxf(s0, -0.5f), 0.5f);
  out[node * 2 + 1] = fminf(fmaxf(s1, -0.5f), 0.5f);
}

// ---------------------------------------------------------------- launch

static inline int cdiv(long a, long b) { return (int)((a + b - 1) / b); }

extern "C" void kernel_launch(void* const* d_in, const int* in_sizes, int n_in,
                              void* d_out, int out_size, void* d_ws, size_t ws_size,
                              hipStream_t stream)
{
  (void)n_in; (void)out_size; (void)ws_size;
  const float* x      = (const float*)d_in[0];
  const int*   ei     = (const int*)  d_in[1];
  const float* w0     = (const float*)d_in[2];
  const float* a_src0 = (const float*)d_in[3];
  const float* a_dst0 = (const float*)d_in[4];
  const float* b0     = (const float*)d_in[5];
  const float* ln0_g  = (const float*)d_in[6];
  const float* ln0_b  = (const float*)d_in[7];
  const float* w1     = (const float*)d_in[8];
  const float* a_src1 = (const float*)d_in[9];
  const float* a_dst1 = (const float*)d_in[10];
  const float* b1     = (const float*)d_in[11];
  const float* ln1_g  = (const float*)d_in[12];
  const float* ln1_b  = (const float*)d_in[13];
  const float* gw_ih  = (const float*)d_in[14];
  /* d_in[15] = gru_w_hh is dead: h0 == 0 */
  const float* gb_ih  = (const float*)d_in[16];
  const float* gb_hh  = (const float*)d_in[17];
  const float* d0w    = (const float*)d_in[18];
  const float* d0b    = (const float*)d_in[19];
  const float* d1w    = (const float*)d_in[20];
  const float* d1b    = (const float*)d_in[21];
  const float* d2w    = (const float*)d_in[22];
  const float* d2b    = (const float*)d_in[23];

  const int n  = in_sizes[0] / 16;   // 50000 (multiple of 16)
  const int e  = in_sizes[1] / 2;    // 800000
  const int et = e + n;              // edges + self loops

  // workspace layout (fp32), with aliasing across phases
  float* ws     = (float*)d_ws;
  float* xp     = ws;                         // n*256   (later: gi [n*192], d0out [n*64])
  float* outacc = xp + (size_t)n * 256;       // n*256   (later: temporal [n*64] + dcat [n*80])
  float* esb    = outacc + (size_t)n * 256;   // n*4
  float* edb    = esb + (size_t)n * 4;        // n*4
  float* segm   = edb + (size_t)n * 4;        // n*4
  float* segs   = segm + (size_t)n * 4;       // n*4
  float* exbuf  = segs + (size_t)n * 4;       // et*4   (later: d1out [n*32])
  float* hbuf   = exbuf + (size_t)et * 4;     // n*64

  const int TB = 256;
  const int mtiles = n / 16;

  // ---- two GAT layers ----
  for (int l = 0; l < 2; ++l) {
    const float* as = (l == 0) ? a_src0 : a_src1;
    const float* ad = (l == 0) ? a_dst0 : a_dst1;
    const float* bb = (l == 0) ? b0 : b1;
    const float* lg = (l == 0) ? ln0_g : ln1_g;
    const float* lb = (l == 0) ? ln0_b : ln1_b;

    long waves = (long)mtiles * 16;           // NB=256 -> 16 col tiles
    if (l == 0)
      gemm_bf16_wmma<16, 256, false><<<cdiv(waves * 32, TB), TB, 0, stream>>>(
          x, w0, nullptr, xp, n, 0);
    else
      gemm_bf16_wmma<64, 256, false><<<cdiv(waves * 32, TB), TB, 0, stream>>>(
          hbuf, w1, nullptr, xp, n, 0);
    node_scores_kernel<<<cdiv((long)n * 4, TB), TB, 0, stream>>>(xp, as, ad, esb, edb, n);
    init_layer_kernel<<<cdiv((long)n * 256, TB), TB, 0, stream>>>(segm, segs, outacc, n);
    edge_max_kernel<<<cdiv(et, TB), TB, 0, stream>>>(ei, esb, edb, segm, e, et);
    edge_exp_kernel<<<cdiv(et, TB), TB, 0, stream>>>(ei, esb, edb, segm, segs, exbuf, e, et);
    edge_agg_kernel<<<cdiv((long)et * 64, TB), TB, 0, stream>>>(ei, xp, exbuf, segs, outacc, e, et);
    headmean_ln_kernel<<<cdiv((long)n * 32, TB), TB, 0, stream>>>(
        outacc, bb, lg, lb, hbuf, n, (l == 0) ? 1 : 0);
  }

  // ---- GRU (single step, zero initial hidden state) ----
  float* gi = xp;                              // reuse xp region: n*192
  {
    long waves = (long)mtiles * 12;            // NB=192
    gemm_bf16_wmma<64, 192, true><<<cdiv(waves * 32, TB), TB, 0, stream>>>(
        hbuf, gw_ih, gb_ih, gi, n, 0);
  }
  float* temporal = outacc;                    // n*64
  gru_kernel<<<cdiv((long)n * 64, TB), TB, 0, stream>>>(gi, gb_hh, temporal, n);

  // ---- MLP decoder ----
  float* dcat = outacc + (size_t)n * 64;       // n*80
  concat_kernel<<<cdiv((long)n * 80, TB), TB, 0, stream>>>(temporal, x, dcat, n);
  float* d0o = xp;                             // n*64 (gi consumed)
  {
    long waves = (long)mtiles * 4;             // NB=64
    gemm_bf16_wmma<80, 64, false><<<cdiv(waves * 32, TB), TB, 0, stream>>>(
        dcat, d0w, d0b, d0o, n, 1);
  }
  float* d1o = exbuf;                          // n*32 (exbuf consumed)
  {
    long waves = (long)mtiles * 2;             // NB=32
    gemm_bf16_wmma<64, 32, false><<<cdiv(waves * 32, TB), TB, 0, stream>>>(
        d0o, d1w, d1b, d1o, n, 1);
  }
  dec2_kernel<<<cdiv(n, TB), TB, 0, stream>>>(d1o, d2w, d2b, (float*)d_out, n);
}
